// FCOSModule_17085379904173
// MI455X (gfx1250) — matrix operand order repaired and checked by reading, not compiled, not run on hardware
//
#include <hip/hip_runtime.h>
#include <hip/hip_bf16.h>

typedef __attribute__((ext_vector_type(16))) _Float16 v16h;
typedef __attribute__((ext_vector_type(8)))  _Float16 v8h;
typedef __attribute__((ext_vector_type(8)))  float    v8f;
typedef __attribute__((ext_vector_type(4)))  float    v4f;
typedef int v4i_vs __attribute__((vector_size(16)));

#define AS1 __attribute__((address_space(1)))
#define AS3 __attribute__((address_space(3)))

#define KTOT 2304   // 256 * 9
#define NCLS 80

#ifndef __has_builtin
#define __has_builtin(x) 0
#endif

// CDNA5 async global->LDS copy path (probe-guarded: falls back to direct
// global loads if the builtin is absent on this toolchain).
#if __has_builtin(__builtin_amdgcn_global_load_async_to_lds_b128)
#define HAVE_ASYNC_LDS 1
#else
#define HAVE_ASYNC_LDS 0
#endif

#if __has_builtin(__builtin_amdgcn_s_wait_asynccnt)
#define WAIT_ASYNC(n) __builtin_amdgcn_s_wait_asynccnt(n)
#else
#define WAIT_ASYNC(n) asm volatile("s_wait_asynccnt %0" :: "i"(n) : "memory")
#endif

__device__ __forceinline__ float sigmoidf_(float x) { return 1.0f / (1.0f + __expf(-x)); }

// ---------------------------------------------------------------------------
// Weight packing: fp32 [.., O, I, 3, 3] -> f16 [O][K], K = (r*3+s)*256 + i
// ---------------------------------------------------------------------------
__global__ void pack_tower_w_kernel(const float* __restrict__ w, _Float16* __restrict__ o16) {
    const size_t total = 4ull * 256 * 256 * 9;
    for (size_t i = blockIdx.x * (size_t)blockDim.x + threadIdx.x; i < total;
         i += (size_t)gridDim.x * blockDim.x) {
        int t = (int)(i % 9);
        size_t r = i / 9;
        int ci = (int)(r & 255); r >>= 8;
        int o  = (int)(r & 255);
        int cv = (int)(r >> 8);
        o16[((size_t)cv * 256 + o) * KTOT + (size_t)t * 256 + ci] = (_Float16)w[i];
    }
}

__global__ void pack_logits_w_kernel(const float* __restrict__ w, _Float16* __restrict__ o16) {
    const size_t total = (size_t)NCLS * 256 * 9;
    for (size_t i = blockIdx.x * (size_t)blockDim.x + threadIdx.x; i < total;
         i += (size_t)gridDim.x * blockDim.x) {
        int t = (int)(i % 9);
        size_t r = i / 9;
        int ci = (int)(r & 255);
        int o  = (int)(r >> 8);
        o16[(size_t)o * KTOT + (size_t)t * 256 + ci] = (_Float16)w[i];
    }
}

// rows 0..3 = bbox, row 4 = ctr, rows 5..15 = zero
__global__ void pack_head16_w_kernel(const float* __restrict__ bboxw,
                                     const float* __restrict__ ctrw,
                                     _Float16* __restrict__ o16) {
    const size_t total = 16ull * KTOT;
    for (size_t i = blockIdx.x * (size_t)blockDim.x + threadIdx.x; i < total;
         i += (size_t)gridDim.x * blockDim.x) {
        int k = (int)(i % KTOT);
        int o = (int)(i / KTOT);
        int t = k / 256, ci = k & 255;
        float v = 0.0f;
        if (o < 4)       v = bboxw[((size_t)o * 256 + ci) * 9 + t];
        else if (o == 4) v = ctrw[(size_t)ci * 9 + t];
        o16[i] = (_Float16)v;
    }
}

__global__ void zero_fill_h_kernel(_Float16* __restrict__ p, int n) {
    int i = blockIdx.x * blockDim.x + threadIdx.x;
    if (i < n) p[i] = (_Float16)0.0f;
}

// ---------------------------------------------------------------------------
// fp32 NCHW [2,256,H,W] -> f16 NHWC [2,H,W,256]
// ---------------------------------------------------------------------------
__global__ void nchw_to_nhwc_kernel(const float* __restrict__ F, _Float16* __restrict__ X,
                                    int H, int Wd) {
    const size_t HW = (size_t)H * Wd;
    const size_t total = 2ull * 256 * HW;
    for (size_t i = blockIdx.x * (size_t)blockDim.x + threadIdx.x; i < total;
         i += (size_t)gridDim.x * blockDim.x) {
        int c    = (int)(i & 255);
        size_t p = (i >> 8) % HW;
        int b    = (int)(i / (HW * 256));
        X[i] = (_Float16)F[((size_t)b * 256 + c) * HW + p];
    }
}

// ---------------------------------------------------------------------------
// Implicit-GEMM 3x3 conv via V_WMMA_F32_16X16X32_F16.
//   X : [2,H,W,256] f16 (NHWC, zero-pad via bounds check / zeroPad buffer)
//   Wk: [Mout][2304] f16, K = tap*256 + c
//   Y : [2,H,W,Mout] f32
// One wave computes a 16(M) x 32(N) tile (two WMMA accumulators sharing the
// A fragment); 72 K-blocks of 32.  Activation tiles are double-buffered in
// per-wave LDS via GLOBAL_LOAD_ASYNC_TO_LDS_B128 when available.
// ---------------------------------------------------------------------------
__global__ __launch_bounds__(128) void conv3x3_wmma_kernel(
    const _Float16* __restrict__ X, const _Float16* __restrict__ Wk,
    const _Float16* __restrict__ zeroPad,
    float* __restrict__ Y, int H, int Wd, int Mout) {
    const int lane  = threadIdx.x;      // 0..31
    const int l16   = lane & 15;
    const int half_ = lane >> 4;
    const int wy    = threadIdx.y;      // wave id in block: 0..3
    const int mTile = blockIdx.y * 4 + wy;
    if (mTile * 16 >= Mout) return;     // uniform per wave

    const int Ntot  = 2 * H * Wd;
    const int nBase = blockIdx.x * 32;

    // Output-column coordinates for the two tiles this lane stores.
    const int n0 = nBase + l16;
    const int n1 = nBase + 16 + l16;

    // A row for this lane: m = mTile*16 + l16 ; halves split K contiguously (k = half*16 + s)
    const _Float16* wrow = Wk + (size_t)(mTile * 16 + l16) * KTOT + half_ * 16;

#if HAVE_ASYNC_LDS
    // [wave][buf][col*32 + k] halves; 2KB per buffer, 16KB per block
    __shared__ __align__(64) _Float16 tile[4][2][32 * 32];
    // Staging assignment: 128 chunks of 16B = 32 cols x 4 chunks; this lane
    // owns chunks t = j*32 + lane  (col = t>>2, q = t&3).
    int sB[4], sY[4], sX[4];
    bool sV[4];
    #pragma unroll
    for (int j = 0; j < 4; ++j) {
        const int t = j * 32 + lane;
        const int c = t >> 2;
        const int n = nBase + c;
        const int nn = (n < Ntot) ? n : (Ntot - 1);
        sV[j] = (n < Ntot);
        sX[j] = nn % Wd;
        sY[j] = (nn / Wd) % H;
        sB[j] = nn / (Wd * H);
    }
    auto stage = [&](int kb, int buf) {
        const int kb0 = kb * 32;
        const int tap = kb0 >> 8;
        const int dy  = tap / 3 - 1;
        const int dx  = tap % 3 - 1;
        const int c0  = kb0 & 255;
        #pragma unroll
        for (int j = 0; j < 4; ++j) {
            const int t = j * 32 + lane;
            const int c = t >> 2, q = t & 3;
            const int ys = sY[j] + dy, xs = sX[j] + dx;
            const bool inb = sV[j] && (unsigned)ys < (unsigned)H && (unsigned)xs < (unsigned)Wd;
            const _Float16* src = inb
                ? X + ((size_t)((sB[j] * H + ys) * Wd + xs) * 256 + c0 + q * 8)
                : zeroPad + q * 8;
            __builtin_amdgcn_global_load_async_to_lds_b128(
                (AS1 v4i_vs*)src,
                (AS3 v4i_vs*)&tile[wy][buf][c * 32 + q * 8],
                0, 0);
        }
    };
    stage(0, 0);
#else
    // Fallback: direct-global fragment loads
    const int nn0 = (n0 < Ntot) ? n0 : (Ntot - 1);
    const int nn1 = (n1 < Ntot) ? n1 : (Ntot - 1);
    const int x0 = nn0 % Wd, y0 = (nn0 / Wd) % H, b0c = nn0 / (Wd * H);
    const int x1 = nn1 % Wd, y1 = (nn1 / Wd) % H, b1c = nn1 / (Wd * H);
#endif

    v8f acc0 = {}, acc1 = {};
    for (int kb = 0; kb < 72; ++kb) {
        const int kb0 = kb * 32;
        __builtin_prefetch(wrow + kb0 + 512, 0, 3);   // global_prefetch_b8 on weight stream
        // --- A fragment (16x32 f16): a[0..15]=K(kb0+half*16..)
        v8h alo = *(const v8h*)(wrow + kb0);
        v8h ahi = *(const v8h*)(wrow + kb0 + 8);
        v16h a;
        #pragma unroll
        for (int i = 0; i < 8; ++i) { a[i] = alo[i]; a[8 + i] = ahi[i]; }

#if HAVE_ASYNC_LDS
        const int buf = kb & 1;
        if (kb < 71) {
            stage(kb + 1, buf ^ 1);
            WAIT_ASYNC(4);          // 4 newer in flight; current buffer done (in-order)
        } else {
            WAIT_ASYNC(0);
        }
        asm volatile("" ::: "memory");
        v16h bv0 = *(const v16h*)&tile[wy][buf][(size_t)l16 * 32 + half_ * 16];
        v16h bv1 = *(const v16h*)&tile[wy][buf][(size_t)(16 + l16) * 32 + half_ * 16];
#else
        const int tap = kb0 >> 8;
        const int dy  = tap / 3 - 1;
        const int dx  = tap % 3 - 1;
        const int c0  = (kb0 & 255) + half_ * 16;
        v16h bv0 = {}, bv1 = {};
        {
            const int ys = y0 + dy, xs = x0 + dx;
            if ((n0 < Ntot) && (unsigned)ys < (unsigned)H && (unsigned)xs < (unsigned)Wd)
                bv0 = *(const v16h*)(X + ((size_t)((b0c * H + ys) * Wd + xs) * 256 + c0));
        }
        {
            const int ys = y1 + dy, xs = x1 + dx;
            if ((n1 < Ntot) && (unsigned)ys < (unsigned)H && (unsigned)xs < (unsigned)Wd)
                bv1 = *(const v16h*)(X + ((size_t)((b1c * H + ys) * Wd + xs) * 256 + c0));
        }
#endif
        acc0 = __builtin_amdgcn_wmma_f32_16x16x32_f16(false, a, false, bv0,
                                                      (short)0, acc0, false, false);
        acc1 = __builtin_amdgcn_wmma_f32_16x16x32_f16(false, a, false, bv1,
                                                      (short)0, acc1, false, false);
    }
    // acc[r] = D[m = half*8 + r][n] -> contiguous channel store
    if (n0 < Ntot) {
        float* yp = Y + (size_t)n0 * Mout + mTile * 16 + half_ * 8;
        v4f lo = {acc0[0], acc0[1], acc0[2], acc0[3]};
        v4f hi = {acc0[4], acc0[5], acc0[6], acc0[7]};
        *(v4f*)yp = lo;
        *(v4f*)(yp + 4) = hi;
    }
    if (n1 < Ntot) {
        float* yp = Y + (size_t)n1 * Mout + mTile * 16 + half_ * 8;
        v4f lo = {acc1[0], acc1[1], acc1[2], acc1[3]};
        v4f hi = {acc1[4], acc1[5], acc1[6], acc1[7]};
        *(v4f*)yp = lo;
        *(v4f*)(yp + 4) = hi;
    }
}

// ---------------------------------------------------------------------------
// GroupNorm statistics: one block per (b, group); 32 groups of 8 channels.
// ---------------------------------------------------------------------------
__global__ __launch_bounds__(256) void gn_stats_kernel(const float* __restrict__ Y,
                                                       float* __restrict__ stats, int HW) {
    const int bg = blockIdx.x;
    const int b = bg >> 5, g = bg & 31;
    const float* base = Y + (size_t)b * HW * 256 + g * 8;
    float s = 0.f, s2 = 0.f;
    for (int p = threadIdx.x; p < HW; p += blockDim.x) {
        const float* q = base + (size_t)p * 256;
        #pragma unroll
        for (int c = 0; c < 8; ++c) { float v = q[c]; s += v; s2 += v * v; }
    }
    __shared__ float ls[256], ls2[256];
    ls[threadIdx.x] = s; ls2[threadIdx.x] = s2;
    __syncthreads();
    for (int off = 128; off > 0; off >>= 1) {
        if ((int)threadIdx.x < off) {
            ls[threadIdx.x]  += ls[threadIdx.x + off];
            ls2[threadIdx.x] += ls2[threadIdx.x + off];
        }
        __syncthreads();
    }
    if (threadIdx.x == 0) {
        float cnt = (float)HW * 8.0f;
        float mu  = ls[0] / cnt;
        float var = ls2[0] / cnt - mu * mu;
        stats[bg * 2]     = mu;
        stats[bg * 2 + 1] = rsqrtf(var + 1e-5f);
    }
}

// (y - mu)*rstd*gamma + beta, ReLU, -> f16 NHWC
__global__ void gn_apply_relu_kernel(const float* __restrict__ Y, const float* __restrict__ stats,
                                     const float* __restrict__ gamma, const float* __restrict__ beta,
                                     _Float16* __restrict__ Xo, int HW) {
    const size_t total = 2ull * HW * 256;
    for (size_t i = blockIdx.x * (size_t)blockDim.x + threadIdx.x; i < total;
         i += (size_t)gridDim.x * blockDim.x) {
        int c = (int)(i & 255);
        int b = (int)(i / ((size_t)HW * 256));
        int g = c >> 3;
        float mu = stats[(b * 32 + g) * 2];
        float rs = stats[(b * 32 + g) * 2 + 1];
        float v = (Y[i] - mu) * rs * gamma[c] + beta[c];
        Xo[i] = (_Float16)(v > 0.0f ? v : 0.0f);
    }
}

// ---------------------------------------------------------------------------
// Decode: scores[b][p*80+c] = sig(logit+bias)*sig(ctr), boxes with clip.
// ---------------------------------------------------------------------------
__global__ void decode_kernel(const float* __restrict__ LG, const float* __restrict__ RC,
                              const float* __restrict__ logits_b, const float* __restrict__ bbox_b,
                              const float* __restrict__ ctr_b, const float* __restrict__ scales,
                              int level, float strideF,
                              float* __restrict__ SC, float* __restrict__ BX,
                              int H, int Wd) {
    const int HW = H * Wd;
    const int total = 2 * HW;
    const float scale = scales[level];
    for (int i = blockIdx.x * blockDim.x + threadIdx.x; i < total;
         i += gridDim.x * blockDim.x) {
        int b = i / HW, p = i - b * HW;
        int px = p % Wd, py = p / Wd;
        const float* rc = RC + (size_t)i * 16;
        float l  = fmaxf((rc[0] + bbox_b[0]) * scale, 0.f) * strideF;
        float tt = fmaxf((rc[1] + bbox_b[1]) * scale, 0.f) * strideF;
        float rr = fmaxf((rc[2] + bbox_b[2]) * scale, 0.f) * strideF;
        float bb = fmaxf((rc[3] + bbox_b[3]) * scale, 0.f) * strideF;
        float cx = px * strideF + strideF * 0.5f;
        float cy = py * strideF + strideF * 0.5f;
        float* bx = BX + (size_t)i * 4;
        bx[0] = fminf(fmaxf(cx - l, 0.f), 1151.f);
        bx[1] = fminf(fmaxf(cy - tt, 0.f), 799.f);
        bx[2] = fminf(fmaxf(cx + rr, 0.f), 1151.f);
        bx[3] = fminf(fmaxf(cy + bb, 0.f), 799.f);
        float cs = sigmoidf_(rc[4] + ctr_b[0]);
        const float* lg = LG + (size_t)i * 80;
        float* sc = SC + (size_t)i * 80;
        for (int c = 0; c < 80; ++c) sc[c] = sigmoidf_(lg[c] + logits_b[c]) * cs;
    }
}

// ---------------------------------------------------------------------------
// Top-1000 per (b, level): histogram threshold -> compact -> bitonic sort ->
// gather boxes into d_out [2, 5000, 5].
// ---------------------------------------------------------------------------
__global__ __launch_bounds__(1024) void topk_kernel(const float* __restrict__ SC,
                                                    const float* __restrict__ BX,
                                                    float* __restrict__ out,
                                                    int HW, int level) {
    const int b = blockIdx.x;
    const int N = HW * 80;
    const float* sc = SC + (size_t)b * N;
    __shared__ unsigned hist[2048];
    __shared__ float svals[1024];
    __shared__ int   sidx[1024];
    __shared__ unsigned cntA, cntB, thrBin, baseA;
    const int tid = threadIdx.x;

    for (int i = tid; i < 2048; i += 1024) hist[i] = 0u;
    __syncthreads();
    for (int i = tid; i < N; i += 1024)
        atomicAdd(&hist[__float_as_uint(sc[i]) >> 21], 1u);
    __syncthreads();
    if (tid == 0) {
        unsigned run = 0; int t = 2047;
        for (; t > 0; --t) {
            unsigned nx = run + hist[t];
            if (nx >= 1000u) break;
            run = nx;
        }
        thrBin = (unsigned)t;
        baseA  = run;
        cntA = 0u; cntB = 0u;
    }
    __syncthreads();
    const unsigned tb = thrBin, ba = baseA;
    for (int i = tid; i < N; i += 1024) {
        float v = sc[i];
        unsigned bin = __float_as_uint(v) >> 21;
        if (bin > tb) {
            unsigned s = atomicAdd(&cntA, 1u);
            svals[s] = v; sidx[s] = i;
        } else if (bin == tb) {
            unsigned s = atomicAdd(&cntB, 1u);
            if (ba + s < 1000u) { svals[ba + s] = v; sidx[ba + s] = i; }
        }
    }
    __syncthreads();
    if (tid >= 1000) { svals[tid] = -1e30f; sidx[tid] = 0; }
    __syncthreads();
    for (int k2 = 2; k2 <= 1024; k2 <<= 1) {
        for (int j = k2 >> 1; j > 0; j >>= 1) {
            int ixj = tid ^ j;
            if (ixj > tid) {
                bool desc = ((tid & k2) == 0);
                float a = svals[tid], c = svals[ixj];
                bool sw = desc ? (a < c) : (a > c);
                if (sw) {
                    svals[tid] = c; svals[ixj] = a;
                    int ti = sidx[tid]; sidx[tid] = sidx[ixj]; sidx[ixj] = ti;
                }
            }
            __syncthreads();
        }
    }
    if (tid < 1000) {
        int idx = sidx[tid];
        int pos = idx / 80;
        const float* bx = BX + ((size_t)b * HW + pos) * 4;
        float* o = out + ((size_t)b * 5000 + (size_t)level * 1000 + tid) * 5;
        o[0] = bx[0]; o[1] = bx[1]; o[2] = bx[2]; o[3] = bx[3]; o[4] = svals[tid];
    }
}

// ---------------------------------------------------------------------------
extern "C" void kernel_launch(void* const* d_in, const int* in_sizes, int n_in,
                              void* d_out, int out_size, void* d_ws, size_t ws_size,
                              hipStream_t stream) {
    (void)in_sizes; (void)n_in; (void)out_size; (void)ws_size;
    static const int LH[5] = {100, 50, 25, 13, 7};
    static const int LW[5] = {144, 72, 36, 18, 9};
    static const int LS[5] = {8, 16, 32, 64, 128};

    const float* feats[5] = {(const float*)d_in[0], (const float*)d_in[1],
                             (const float*)d_in[2], (const float*)d_in[3],
                             (const float*)d_in[4]};
    const float* cls_tw   = (const float*)d_in[5];
    const float* cls_gn_g = (const float*)d_in[7];
    const float* cls_gn_b = (const float*)d_in[8];
    const float* box_tw   = (const float*)d_in[9];
    const float* box_gn_g = (const float*)d_in[11];
    const float* box_gn_b = (const float*)d_in[12];
    const float* logits_w = (const float*)d_in[13];
    const float* logits_b = (const float*)d_in[14];
    const float* bbox_w   = (const float*)d_in[15];
    const float* bbox_b   = (const float*)d_in[16];
    const float* ctr_w    = (const float*)d_in[17];
    const float* ctr_b    = (const float*)d_in[18];
    const float* scales   = (const float*)d_in[19];
    float* outp = (float*)d_out;

    // -------- workspace layout (level-0 maxima, reused across levels) --------
    char* ws = (char*)d_ws;
    size_t cur = 0;
    auto alloc = [&](size_t bytes) -> char* {
        cur = (cur + 255) & ~(size_t)255;
        char* p = ws + cur;
        cur += bytes;
        return p;
    };
    const size_t HWmax = 14400;
    _Float16* wCls = (_Float16*)alloc(4ull * 256 * KTOT * 2);
    _Float16* wBox = (_Float16*)alloc(4ull * 256 * KTOT * 2);
    _Float16* wLog = (_Float16*)alloc((size_t)NCLS * KTOT * 2);
    _Float16* wHd  = (_Float16*)alloc(16ull * KTOT * 2);
    _Float16* zeroPad = (_Float16*)alloc(2048ull * 2);
    _Float16* XA = (_Float16*)alloc(2ull * HWmax * 256 * 2);
    _Float16* XB = (_Float16*)alloc(2ull * HWmax * 256 * 2);
    _Float16* CT = (_Float16*)alloc(2ull * HWmax * 256 * 2);
    _Float16* BT = (_Float16*)alloc(2ull * HWmax * 256 * 2);
    float* YF   = (float*)alloc(2ull * HWmax * 256 * 4);
    float* LOGb = (float*)alloc(2ull * HWmax * NCLS * 4);
    float* RC   = (float*)alloc(2ull * HWmax * 16 * 4);
    float* SC   = (float*)alloc(2ull * HWmax * NCLS * 4);
    float* BXb  = (float*)alloc(2ull * HWmax * 4 * 4);
    float* stats = (float*)alloc(64ull * 2 * 4);

    auto cdiv = [](long long a, long long b) -> int { return (int)((a + b - 1) / b); };

    // -------- pack weights to f16 GEMM layout, zero the OOB pad --------
    pack_tower_w_kernel<<<cdiv(4ll * 256 * 256 * 9, 256), 256, 0, stream>>>(cls_tw, wCls);
    pack_tower_w_kernel<<<cdiv(4ll * 256 * 256 * 9, 256), 256, 0, stream>>>(box_tw, wBox);
    pack_logits_w_kernel<<<cdiv((long long)NCLS * 256 * 9, 256), 256, 0, stream>>>(logits_w, wLog);
    pack_head16_w_kernel<<<cdiv(16ll * KTOT, 256), 256, 0, stream>>>(bbox_w, ctr_w, wHd);
    zero_fill_h_kernel<<<8, 256, 0, stream>>>(zeroPad, 2048);

    for (int lvl = 0; lvl < 5; ++lvl) {
        const int H = LH[lvl], Wd = LW[lvl];
        const int HW = H * Wd;
        const int Ntot = 2 * HW;
        const float strideF = (float)LS[lvl];

        nchw_to_nhwc_kernel<<<cdiv(2ll * HW * 256, 256), 256, 0, stream>>>(feats[lvl], XA, H, Wd);

        dim3 cblk(32, 4);
        auto convL = [&](const _Float16* Xi, const _Float16* Wp, float* Yo, int Mout) {
            dim3 cgrd(cdiv(Ntot, 32), cdiv(Mout / 16, 4));
            conv3x3_wmma_kernel<<<cgrd, cblk, 0, stream>>>(Xi, Wp, zeroPad, Yo, H, Wd, Mout);
        };
        auto gnrelu = [&](const float* g, const float* be, _Float16* Xo) {
            gn_stats_kernel<<<64, 256, 0, stream>>>(YF, stats, HW);
            gn_apply_relu_kernel<<<cdiv(2ll * HW * 256, 256), 256, 0, stream>>>(
                YF, stats, g, be, Xo, HW);
        };

        // cls tower: XA -> XB -> CT -> XB -> CT
        {
            const _Float16* in = XA;
            _Float16* outs[4] = {XB, CT, XB, CT};
            for (int i = 0; i < 4; ++i) {
                convL(in, wCls + (size_t)i * 256 * KTOT, YF, 256);
                gnrelu(cls_gn_g + i * 256, cls_gn_b + i * 256, outs[i]);
                in = outs[i];
            }
        }
        // box tower: XA -> XB -> BT -> XB -> BT  (CT preserved)
        {
            const _Float16* in = XA;
            _Float16* outs[4] = {XB, BT, XB, BT};
            for (int i = 0; i < 4; ++i) {
                convL(in, wBox + (size_t)i * 256 * KTOT, YF, 256);
                gnrelu(box_gn_g + i * 256, box_gn_b + i * 256, outs[i]);
                in = outs[i];
            }
        }
        // heads
        convL(CT, wLog, LOGb, NCLS);   // class logits (bias added in decode)
        convL(BT, wHd, RC, 16);        // rows 0..3 reg, 4 ctr

        decode_kernel<<<cdiv(2ll * HW, 128), 128, 0, stream>>>(
            LOGb, RC, logits_b, bbox_b, ctr_b, scales, lvl, strideF, SC, BXb, H, Wd);

        topk_kernel<<<2, 1024, 0, stream>>>(SC, BXb, outp, HW, lvl);
    }
}
